// MessageLayer_16939351015861
// MI455X (gfx1250) — matrix operand order repaired and checked by reading, not compiled, not run on hardware
//
#include <hip/hip_runtime.h>

#define N_NODES 10000
#define N_EDGES 640000
#define HID 128

// ---------- CDNA5 WMMA types ----------
typedef __attribute__((ext_vector_type(16))) __bf16 v16bf;
typedef __attribute__((ext_vector_type(8)))  float  v8f;
typedef __attribute__((ext_vector_type(4)))  unsigned int u32x4;

union Frag { u32x4 u[2]; v16bf v; };   // 32 bytes = 16 bf16 = one A/B operand

__device__ __forceinline__ v8f wmma_bf16(v16bf a, v16bf b, v8f c) {
  // D = A(16x32 bf16) * B(32x16 bf16) + C(16x16 f32)
  return __builtin_amdgcn_wmma_f32_16x16x32_bf16(false, a, false, b,
                                                 (short)0, c, false, false);
}

__device__ __forceinline__ unsigned short f2bf(float f) {
  unsigned int u = __builtin_bit_cast(unsigned int, f);
  u += 0x7FFFu + ((u >> 16) & 1u);          // round-to-nearest-even
  return (unsigned short)(u >> 16);
}

__device__ __forceinline__ v8f vzero() {
  v8f z = {0.f,0.f,0.f,0.f,0.f,0.f,0.f,0.f};
  return z;
}

// ---------- prep kernels ----------
__global__ void k_f32_to_bf16(const float* __restrict__ s,
                              unsigned short* __restrict__ d, int n) {
  int i = blockIdx.x * blockDim.x + threadIdx.x;
  if (i < n) d[i] = f2bf(s[i]);
}

__global__ void k_zero_f32(float* __restrict__ p, int n) {
  int i = blockIdx.x * blockDim.x + threadIdx.x;
  if (i < n) p[i] = 0.f;
}

__global__ void k_copy_f32(const float* __restrict__ s, float* __restrict__ d, int n) {
  int i = blockIdx.x * blockDim.x + threadIdx.x;
  if (i < n) d[i] = s[i];
}

// Pre-swizzle a [K x 128] f32 weight into per-wave B-fragments (bf16):
// out[((nt*KC + kc)*32 + lane)*16 + j] = w[kc*32 + (j/8)*16 + (lane/16)*8 + j%8][nt*16 + lane%16]
__global__ void k_pack_w(const float* __restrict__ w,
                         unsigned short* __restrict__ out, int K) {
  int idx = blockIdx.x * blockDim.x + threadIdx.x;
  int total = K * HID;
  if (idx >= total) return;
  int KC   = K >> 5;
  int j    = idx & 15;
  int lane = (idx >> 4) & 31;
  int rest = idx >> 9;
  int kc   = rest % KC;
  int nt   = rest / KC;
  int k = kc * 32 + (j >> 3) * 16 + (lane >> 4) * 8 + (j & 7);
  int n = nt * 16 + (lane & 15);
  out[idx] = f2bf(w[k * HID + n]);
}

// ---------- edge message kernel: 1 wave = 32 edges (M=32), 2 waves/block ----------
__global__ __launch_bounds__(64)
void edge_msg_kernel(const long long* __restrict__ ei,
                     const float* __restrict__ eattr,
                     const unsigned short* __restrict__ hB,
                     const unsigned short* __restrict__ w1p,
                     const float* __restrict__ w1r256,
                     const float* __restrict__ b1,
                     const unsigned short* __restrict__ w2p,
                     const float* __restrict__ b2,
                     float* __restrict__ agg) {
  __shared__ unsigned short sH[2 * 32 * HID];   // 16 KB hidden staging (8 KB/wave)
  __shared__ unsigned short sW2[HID * HID];     // 32 KB layer-2 weights (bf16 fragments)

  const int tid  = threadIdx.x;
  const int wave = tid >> 5;
  const int lane = tid & 31;
  const int half = lane >> 4;
  const int ln   = lane & 15;

  // ---- cooperative stage of packed W2 into LDS (one barrier, both waves) ----
  {
    u32x4* d = (u32x4*)sW2;
    const u32x4* s = (const u32x4*)(const void*)w2p;
    #pragma unroll
    for (int i = 0; i < (HID * HID / 8) / 64; ++i)   // 2048 vec4 / 64 thr = 32
      d[tid + i * 64] = s[tid + i * 64];
  }
  __syncthreads();

  const long tile  = (long)blockIdx.x * 2 + wave;     // 20000 tiles of 32 edges
  const long eBase = tile * 32;

  const long long* srcI = ei;
  const long long* dstI = ei + (long)N_EDGES;

  long rowS[2], rowD[2];
#pragma unroll
  for (int t = 0; t < 2; ++t) {
    rowS[t] = (long)srcI[eBase + t * 16 + ln];
    rowD[t] = (long)dstI[eBase + t * 16 + ln];
  }

  v8f acc[2][8];
#pragma unroll
  for (int t = 0; t < 2; ++t)
#pragma unroll
    for (int nt = 0; nt < 8; ++nt) acc[t][nt] = vzero();

  auto loadA1 = [&](Frag& f, int t, int kc) {
    const unsigned short* abase = (kc < 4) ? (hB + rowS[t] * HID + kc * 32)
                                           : (hB + rowD[t] * HID + (kc - 4) * 32);
    f.u[0] = *(const u32x4*)(const void*)(abase + half * 8);
    f.u[1] = *(const u32x4*)(const void*)(abase + 16 + half * 8);
  };
  auto loadB1 = [&](Frag& f, int kc, int nt) {
    const unsigned short* wb = w1p + (((nt * 8 + kc) * 32) + lane) * 16;
    f.u[0] = *(const u32x4*)(const void*)(wb);
    f.u[1] = *(const u32x4*)(const void*)(wb + 8);
  };

  // ---- layer 1: K=256, two M-tiles share each B fragment ----
  Frag a[2][2];            // [buffer][tile]
  Frag br[4];
  loadA1(a[0][0], 0, 0);
  loadA1(a[0][1], 1, 0);
  loadB1(br[0], 0, 0);
  loadB1(br[1], 0, 1);
  loadB1(br[2], 0, 2);
#pragma unroll
  for (int kc = 0; kc < 8; ++kc) {
    const int cb = kc & 1;
    if (kc < 7) {
      loadA1(a[cb ^ 1][0], 0, kc + 1);
      loadA1(a[cb ^ 1][1], 1, kc + 1);
    }
#pragma unroll
    for (int nt = 0; nt < 8; ++nt) {
      const int step = kc * 8 + nt;
      const int pf   = step + 3;
      if (pf < 64) loadB1(br[pf & 3], pf >> 3, pf & 7);
      acc[0][nt] = wmma_bf16(a[cb][0].v, br[step & 3].v, acc[0][nt]);
      acc[1][nt] = wmma_bf16(a[cb][1].v, br[step & 3].v, acc[1][nt]);
    }
  }

  // ---- epilogue 1: + bias + edge_attr * w1[256,:], ReLU, stage to LDS ----
  unsigned short* myH = sH + wave * 32 * HID;
#pragma unroll
  for (int t = 0; t < 2; ++t) {
    float ea[8];
#pragma unroll
    for (int v = 0; v < 8; ++v) ea[v] = eattr[eBase + t * 16 + half * 8 + v];
#pragma unroll
    for (int nt = 0; nt < 8; ++nt) {
      const int n = nt * 16 + ln;
      const float bias = b1[n];
      const float wr   = w1r256[n];
#pragma unroll
      for (int v = 0; v < 8; ++v) {
        float x = acc[t][nt][v] + bias + ea[v] * wr;
        x = fmaxf(x, 0.f);
        myH[(t * 16 + half * 8 + v) * HID + n] = f2bf(x);  // m = half*8+v (C/D layout)
      }
    }
  }

  auto loadA2 = [&](Frag& f, int t, int kc) {
    const unsigned short* abase = myH + (t * 16 + ln) * HID + kc * 32;
    f.u[0] = *(const u32x4*)(const void*)(abase + half * 8);
    f.u[1] = *(const u32x4*)(const void*)(abase + 16 + half * 8);
  };
  auto loadB2 = [&](Frag& f, int kc, int nt) {
    const unsigned short* wb = sW2 + (((nt * 4 + kc) * 32) + lane) * 16;
    f.u[0] = *(const u32x4*)(const void*)(wb);
    f.u[1] = *(const u32x4*)(const void*)(wb + 8);
  };

  // ---- layer 2: K=128, all operands from LDS, same pipeline ----
#pragma unroll
  for (int t = 0; t < 2; ++t)
#pragma unroll
    for (int nt = 0; nt < 8; ++nt) acc[t][nt] = vzero();
  loadA2(a[0][0], 0, 0);
  loadA2(a[0][1], 1, 0);
  loadB2(br[0], 0, 0);
  loadB2(br[1], 0, 1);
  loadB2(br[2], 0, 2);
#pragma unroll
  for (int kc = 0; kc < 4; ++kc) {
    const int cb = kc & 1;
    if (kc < 3) {
      loadA2(a[cb ^ 1][0], 0, kc + 1);
      loadA2(a[cb ^ 1][1], 1, kc + 1);
    }
#pragma unroll
    for (int nt = 0; nt < 8; ++nt) {
      const int step = kc * 8 + nt;
      const int pf   = step + 3;
      if (pf < 32) loadB2(br[pf & 3], pf >> 3, pf & 7);
      acc[0][nt] = wmma_bf16(a[cb][0].v, br[step & 3].v, acc[0][nt]);
      acc[1][nt] = wmma_bf16(a[cb][1].v, br[step & 3].v, acc[1][nt]);
    }
  }

  // ---- scatter-add into agg via hardware f32 atomics ----
#pragma unroll
  for (int t = 0; t < 2; ++t) {
    long dstRow[8];
#pragma unroll
    for (int v = 0; v < 8; ++v)
      dstRow[v] = (long)dstI[eBase + t * 16 + half * 8 + v];
#pragma unroll
    for (int nt = 0; nt < 8; ++nt) {
      const int n = nt * 16 + ln;
      const float bias = b2[n];
#pragma unroll
      for (int v = 0; v < 8; ++v) {
        float x = acc[t][nt][v] + bias;
        unsafeAtomicAdd(&agg[dstRow[v] * HID + n], x);
      }
    }
  }
}

// ---------- node update kernel: 1 wave = 16 nodes, 4 waves/block ----------
__global__ __launch_bounds__(128)
void node_upd_kernel(const float* __restrict__ h,
                     const unsigned short* __restrict__ hB,
                     const unsigned short* __restrict__ aggB,
                     const unsigned short* __restrict__ uw1p,
                     const float* __restrict__ ub1,
                     const unsigned short* __restrict__ uw2p,
                     const float* __restrict__ ub2,
                     float* __restrict__ out) {
  __shared__ unsigned short sH[4 * 16 * HID];
  __shared__ unsigned short sW2[HID * HID];

  const int tid  = threadIdx.x;
  const int wave = tid >> 5;
  const int lane = tid & 31;
  const int half = lane >> 4;
  const int ln   = lane & 15;

  // stage W2 before any wave exits (barrier executed by every wave)
  {
    u32x4* d = (u32x4*)sW2;
    const u32x4* s = (const u32x4*)(const void*)uw2p;
    #pragma unroll
    for (int i = 0; i < (HID * HID / 8) / 128; ++i)
      d[tid + i * 128] = s[tid + i * 128];
  }
  __syncthreads();

  const long tile = (long)blockIdx.x * 4 + wave;
  if (tile >= (N_NODES / 16)) return;   // wave-uniform exit (EXEC stays all-1)
  const long nBase = tile * 16;
  const long rowA  = nBase + ln;

  v8f acc[8];
#pragma unroll
  for (int nt = 0; nt < 8; ++nt) acc[nt] = vzero();

  auto loadA1 = [&](Frag& f, int kc) {
    const unsigned short* abase = (kc < 4) ? (hB + rowA * HID + kc * 32)
                                           : (aggB + rowA * HID + (kc - 4) * 32);
    f.u[0] = *(const u32x4*)(const void*)(abase + half * 8);
    f.u[1] = *(const u32x4*)(const void*)(abase + 16 + half * 8);
  };
  auto loadB1 = [&](Frag& f, int kc, int nt) {
    const unsigned short* wb = uw1p + (((nt * 8 + kc) * 32) + lane) * 16;
    f.u[0] = *(const u32x4*)(const void*)(wb);
    f.u[1] = *(const u32x4*)(const void*)(wb + 8);
  };

  // ---- layer 1: K = 256 (h || agg) ----
  Frag a0, a1;
  Frag br[4];
  loadA1(a0, 0);
  loadB1(br[0], 0, 0);
  loadB1(br[1], 0, 1);
  loadB1(br[2], 0, 2);
#pragma unroll
  for (int kc = 0; kc < 8; ++kc) {
    Frag& acur = (kc & 1) ? a1 : a0;
    Frag& anxt = (kc & 1) ? a0 : a1;
    if (kc < 7) loadA1(anxt, kc + 1);
#pragma unroll
    for (int nt = 0; nt < 8; ++nt) {
      const int step = kc * 8 + nt;
      const int pf   = step + 3;
      if (pf < 64) loadB1(br[pf & 3], pf >> 3, pf & 7);
      acc[nt] = wmma_bf16(acur.v, br[step & 3].v, acc[nt]);
    }
  }

  unsigned short* myH = sH + wave * 16 * HID;
#pragma unroll
  for (int nt = 0; nt < 8; ++nt) {
    const int n = nt * 16 + ln;
    const float bias = ub1[n];
#pragma unroll
    for (int v = 0; v < 8; ++v) {
      float x = fmaxf(acc[nt][v] + bias, 0.f);
      myH[(half * 8 + v) * HID + n] = f2bf(x);
    }
  }

  auto loadA2 = [&](Frag& f, int kc) {
    const unsigned short* abase = myH + ln * HID + kc * 32;
    f.u[0] = *(const u32x4*)(const void*)(abase + half * 8);
    f.u[1] = *(const u32x4*)(const void*)(abase + 16 + half * 8);
  };
  auto loadB2 = [&](Frag& f, int kc, int nt) {
    const unsigned short* wb = sW2 + (((nt * 4 + kc) * 32) + lane) * 16;
    f.u[0] = *(const u32x4*)(const void*)(wb);
    f.u[1] = *(const u32x4*)(const void*)(wb + 8);
  };

  // ---- layer 2 ----
#pragma unroll
  for (int nt = 0; nt < 8; ++nt) acc[nt] = vzero();
  loadA2(a0, 0);
  loadB2(br[0], 0, 0);
  loadB2(br[1], 0, 1);
  loadB2(br[2], 0, 2);
#pragma unroll
  for (int kc = 0; kc < 4; ++kc) {
    Frag& acur = (kc & 1) ? a1 : a0;
    Frag& anxt = (kc & 1) ? a0 : a1;
    if (kc < 3) loadA2(anxt, kc + 1);
#pragma unroll
    for (int nt = 0; nt < 8; ++nt) {
      const int step = kc * 8 + nt;
      const int pf   = step + 3;
      if (pf < 32) loadB2(br[pf & 3], pf >> 3, pf & 7);
      acc[nt] = wmma_bf16(acur.v, br[step & 3].v, acc[nt]);
    }
  }

  // ---- residual + bias + store ----
#pragma unroll
  for (int nt = 0; nt < 8; ++nt) {
    const int n = nt * 16 + ln;
    const float bias = ub2[n];
#pragma unroll
    for (int v = 0; v < 8; ++v) {
      const long idx = (nBase + half * 8 + v) * HID + n;
      out[idx] = h[idx] + acc[nt][v] + bias;
    }
  }
}

// ---------- launch ----------
extern "C" void kernel_launch(void* const* d_in, const int* in_sizes, int n_in,
                              void* d_out, int out_size, void* d_ws, size_t ws_size,
                              hipStream_t stream) {
  const float*     h      = (const float*)d_in[0];
  const long long* ei     = (const long long*)d_in[1];   // int64 [2, E]
  const float*     eattr  = (const float*)d_in[2];
  const float*     msg_w1 = (const float*)d_in[3];       // [257,128]
  const float*     msg_b1 = (const float*)d_in[4];
  const float*     msg_w2 = (const float*)d_in[5];       // [128,128]
  const float*     msg_b2 = (const float*)d_in[6];
  const float*     upd_w1 = (const float*)d_in[7];       // [256,128]
  const float*     upd_b1 = (const float*)d_in[8];
  const float*     upd_w2 = (const float*)d_in[9];       // [128,128]
  const float*     upd_b2 = (const float*)d_in[10];
  float* out = (float*)d_out;
  (void)in_sizes; (void)n_in; (void)out_size; (void)ws_size;

  char* ws = (char*)d_ws;
  size_t off = 0;
  auto carve = [&](size_t bytes) -> void* {
    void* p = ws + off;
    off += (bytes + 255) & ~(size_t)255;
    return p;
  };
  unsigned short* hB   = (unsigned short*)carve((size_t)N_NODES * HID * 2);
  float*          aggF = (float*)         carve((size_t)N_NODES * HID * 4);
  unsigned short* aggB = (unsigned short*)carve((size_t)N_NODES * HID * 2);
  unsigned short* w1p  = (unsigned short*)carve((size_t)256 * HID * 2);
  float*          w1r  = (float*)         carve((size_t)HID * 4);
  unsigned short* w2p  = (unsigned short*)carve((size_t)HID * HID * 2);
  unsigned short* uw1p = (unsigned short*)carve((size_t)256 * HID * 2);
  unsigned short* uw2p = (unsigned short*)carve((size_t)HID * HID * 2);

  const int nh = N_NODES * HID;                 // 1,280,000
  // prep
  k_f32_to_bf16<<<(nh + 255) / 256, 256, 0, stream>>>(h, hB, nh);
  k_zero_f32  <<<(nh + 255) / 256, 256, 0, stream>>>(aggF, nh);
  k_pack_w    <<<(256 * HID + 255) / 256, 256, 0, stream>>>(msg_w1, w1p, 256);
  k_copy_f32  <<<1, 128, 0, stream>>>(msg_w1 + 256 * HID, w1r, HID);
  k_pack_w    <<<(HID * HID + 255) / 256, 256, 0, stream>>>(msg_w2, w2p, HID);
  k_pack_w    <<<(256 * HID + 255) / 256, 256, 0, stream>>>(upd_w1, uw1p, 256);
  k_pack_w    <<<(HID * HID + 255) / 256, 256, 0, stream>>>(upd_w2, uw2p, HID);

  // edge messages + scatter: 32 edges/wave * 2 waves/block -> 10000 blocks
  edge_msg_kernel<<<N_EDGES / 64, 64, 0, stream>>>(ei, eattr, hB, w1p, w1r,
                                                   msg_b1, w2p, msg_b2, aggF);

  k_f32_to_bf16<<<(nh + 255) / 256, 256, 0, stream>>>(aggF, aggB, nh);

  // node update: 625 tiles / 4 waves per block
  node_upd_kernel<<<(N_NODES / 16 + 3) / 4, 128, 0, stream>>>(h, hB, aggB,
                                                              uw1p, upd_b1,
                                                              uw2p, upd_b2, out);
}
